// BernoulliGatedChannelStack_13073880449225
// MI455X (gfx1250) — matrix operand order repaired and checked by reading, not compiled, not run on hardware
//
#include <hip/hip_runtime.h>
#include <hip/hip_bf16.h>

// Problem sizes (fixed by the reference).
#define B_  16384
#define F_  512
#define N_  32
#define C_  64
#define NC_ 2048   // N_*C_

typedef __attribute__((ext_vector_type(16))) __bf16 v16bf;
typedef __attribute__((ext_vector_type(8)))  __bf16 v8bf;
typedef __attribute__((ext_vector_type(8)))  float  v8f;

// ---------- kernel 1: gates + normalization scales ----------
// One wave32 per row b; lane = gate index n.
// scale[b,n] = G[b,n] / sum(G[b,:])  (0 if sum==0)  -- the C factors cancel.
__global__ __launch_bounds__(256) void gate_scale_kernel(
    const float* __restrict__ x, const float* __restrict__ Wg,
    float* __restrict__ Gout, float* __restrict__ scale) {
    const int wave = threadIdx.x >> 5;
    const int lane = threadIdx.x & 31;
    const int b = blockIdx.x * 8 + wave;

    const float* xr = x + (size_t)b * F_;
    float acc = 0.0f;
#pragma unroll 8
    for (int f = 0; f < F_; ++f)
        acc += xr[f] * Wg[f * N_ + lane];

    const float p = 1.0f / (1.0f + __expf(-acc));

    // deterministic counter-based uniform draw (stands in for threefry(key=42))
    unsigned idx = (unsigned)(b * N_ + lane);
    unsigned s = idx * 0x9E3779B9u + 42u * 0x7F4A7C15u;
    s ^= s >> 16; s *= 0x85EBCA6Bu;
    s ^= s >> 13; s *= 0xC2B2AE35u;
    s ^= s >> 16;
    const float u = (float)(s >> 8) * (1.0f / 16777216.0f);
    const float g = (u < p) ? 1.0f : 0.0f;

    float gs = g;
#pragma unroll
    for (int off = 16; off > 0; off >>= 1)
        gs += __shfl_xor(gs, off, 32);

    Gout[idx] = g;
    scale[idx] = (gs > 0.0f) ? (g / gs) : 0.0f;
}

// ---------- kernel 2: tiled bf16 WMMA GEMM with fused gate-scale epilogue ----------
// Block tile 64(M) x 128(N_out); 8 waves in 2(M) x 4(N); wave tile 32x32 (2x2 WMMA).
// K staged 64-deep in LDS (two 32-deep WMMA chunks per barrier pair).
#define KS   64   // K per LDS stage
#define LDA  72   // bf16 elements per A row in LDS (64 + pad)
#define LDBT 72   // bf16 elements per B column in LDS (64 + pad)

__global__ __launch_bounds__(256) void gemm_gated_kernel(
    const float* __restrict__ x, const float* __restrict__ Wc,
    const float* __restrict__ scale, float* __restrict__ Y) {
    __shared__ __align__(16) unsigned short sA[64 * LDA];    // [row][k]
    __shared__ __align__(16) unsigned short sB[128 * LDBT];  // [col][k] (transposed)
    __shared__ float sS[64 * 2];                             // per-row scales, 2 n-groups

    const int tid  = threadIdx.x;
    const int lane = tid & 31;
    const int wave = tid >> 5;
    const int wm   = wave & 1;       // M group (x32)
    const int wn   = wave >> 1;      // N group (x32)
    const int half = lane >> 4;
    const int l16  = lane & 15;

    const int M0 = blockIdx.x * 64;
    const int N0 = blockIdx.y * 128;
    const int nBase = N0 >> 6;       // first gate index covered by this block

    if (tid < 128) {
        int row = tid >> 1, w = tid & 1;
        sS[row * 2 + w] = scale[(size_t)(M0 + row) * N_ + nBase + w];
    }

    v8f acc[2][2] = {};

    // A staging: thread -> (row, 16 consecutive k)
    const int aRow = tid >> 2;
    const int aQ   = (tid & 3) * 16;
    // B staging: thread -> (column, 32 consecutive k), gathered down a Wc column
    const int bCol = tid & 127;
    const int bKh  = (tid >> 7) * 32;
    const int bN   = (N0 + bCol) >> 6;
    const int bC   = (N0 + bCol) & 63;
    const float* wcBase = Wc + (size_t)bN * F_ * C_ + bC;

    for (int k0 = 0; k0 < F_; k0 += KS) {
        // ---- A tile: 64x64 f32 -> bf16 LDS (native cvt via __bf16 casts)
        {
            const float* src = x + (size_t)(M0 + aRow) * F_ + k0 + aQ;
#pragma unroll
            for (int g = 0; g < 2; ++g) {
                float4 v0 = *(const float4*)(src + g * 8);
                float4 v1 = *(const float4*)(src + g * 8 + 4);
                v8bf pk;
                pk[0] = (__bf16)v0.x; pk[1] = (__bf16)v0.y;
                pk[2] = (__bf16)v0.z; pk[3] = (__bf16)v0.w;
                pk[4] = (__bf16)v1.x; pk[5] = (__bf16)v1.y;
                pk[6] = (__bf16)v1.z; pk[7] = (__bf16)v1.w;
                *(v8bf*)&sA[aRow * LDA + aQ + g * 8] = pk;
            }
        }
        // ---- B tile: 64x128 f32 (strided) -> bf16 LDS, transposed to [col][k]
        {
            const float* src = wcBase + (size_t)(k0 + bKh) * C_;
#pragma unroll
            for (int g = 0; g < 4; ++g) {
                v8bf pk;
#pragma unroll
                for (int k = 0; k < 8; ++k)
                    pk[k] = (__bf16)src[(size_t)(g * 8 + k) * C_];
                *(v8bf*)&sB[bCol * LDBT + bKh + g * 8] = pk;
            }
        }
        // prefetch next stage's source lines into cache (global_prefetch_b8)
        if (k0 + KS < F_) {
            __builtin_prefetch(x + (size_t)(M0 + aRow) * F_ + k0 + KS + aQ, 0, 3);
            __builtin_prefetch(wcBase + (size_t)(k0 + KS + bKh) * C_, 0, 3);
        }
        __syncthreads();

        // ---- two K=32 WMMA chunks per stage
#pragma unroll
        for (int kc = 0; kc < 2; ++kc) {
            const int kOff = kc * 32;
            v16bf afr[2], bfr[2];
#pragma unroll
            for (int i = 0; i < 2; ++i) {
                const int row = wm * 32 + i * 16 + l16;
                const v8bf lo = *(const v8bf*)&sA[row * LDA + kOff + half * 8];       // K 0..7 / 8..15
                const v8bf hi = *(const v8bf*)&sA[row * LDA + kOff + 16 + half * 8];  // K 16..23 / 24..31
#pragma unroll
                for (int e = 0; e < 8; ++e) { afr[i][e] = lo[e]; afr[i][e + 8] = hi[e]; }
            }
#pragma unroll
            for (int j = 0; j < 2; ++j) {
                const int col = wn * 32 + j * 16 + l16;
                const v8bf lo = *(const v8bf*)&sB[col * LDBT + kOff + half * 8];       // K 0..7 / 8..15
                const v8bf hi = *(const v8bf*)&sB[col * LDBT + kOff + 16 + half * 8];  // K 16..23 / 24..31
#pragma unroll
                for (int e = 0; e < 8; ++e) { bfr[j][e] = lo[e]; bfr[j][e + 8] = hi[e]; }
            }
#pragma unroll
            for (int i = 0; i < 2; ++i)
#pragma unroll
                for (int j = 0; j < 2; ++j)
                    acc[i][j] = __builtin_amdgcn_wmma_f32_16x16x32_bf16(
                        false, afr[i], false, bfr[j], (short)0, acc[i][j], false, false);
        }
        __syncthreads();
    }

    // ---- epilogue: scale by gate/normalizer and store
#pragma unroll
    for (int i = 0; i < 2; ++i) {
#pragma unroll
        for (int j = 0; j < 2; ++j) {
            const int colBase = wn * 32 + j * 16;
            const int ng  = colBase >> 6;           // which n-group (0/1) in this block
            const int col = N0 + colBase + l16;
#pragma unroll
            for (int r = 0; r < 8; ++r) {
                const int mLocal = wm * 32 + i * 16 + r + 8 * half;  // C/D layout: M = r + 8*(lane/16)
                const float s = sS[mLocal * 2 + ng];
                Y[(size_t)(M0 + mLocal) * NC_ + col] = acc[i][j][r] * s;
            }
        }
    }
}

extern "C" void kernel_launch(void* const* d_in, const int* in_sizes, int n_in,
                              void* d_out, int out_size, void* d_ws, size_t ws_size,
                              hipStream_t stream) {
    const float* x  = (const float*)d_in[0];
    const float* Wg = (const float*)d_in[1];
    const float* Wc = (const float*)d_in[2];

    float* Y = (float*)d_out;                   // [B, N*C]
    float* G = Y + (size_t)B_ * NC_;            // [B, N] appended (tuple output order)
    float* scale = (float*)d_ws;                // [B, N] f32 scratch (2 MB)

    gate_scale_kernel<<<B_ / 8, 256, 0, stream>>>(x, Wg, G, scale);
    gemm_gated_kernel<<<dim3(B_ / 64, NC_ / 128), 256, 0, stream>>>(x, Wc, scale, Y);
}
